// AVETransformersCRF_89223650607581
// MI455X (gfx1250) — matrix operand order repaired and checked by reading, not compiled, not run on hardware
//
#include <hip/hip_runtime.h>
#include <hip/hip_bf16.h>
#include <math.h>

typedef __bf16 bf16_t;
typedef __attribute__((ext_vector_type(16))) __bf16 v16bf;
typedef __attribute__((ext_vector_type(8)))  float  v8f;

#define BB 32
#define SS 128
#define TT 192
#define SA 32
#define TA 64
#define DD 768
#define HDIM 384
#define GG 1536
#define LLAB 30
#define START_I 27
#define END_I 28
#define HPAD 392   // padded LDS row stride for h (bf16 elems), even

__device__ __forceinline__ float sigmf_(float x){ return 1.0f/(1.0f+__expf(-x)); }

__device__ __forceinline__ v8f zero8(){
  v8f z;
#pragma unroll
  for (int i=0;i<8;++i) z[i]=0.0f;
  return z;
}

// A fragment: 16x32 bf16, row-major source A[row][k], lda in elements.
__device__ __forceinline__ v16bf load_a_frag(const bf16_t* A, int lda, int m0, int k0){
  int lane = threadIdx.x & 31;
  int half = lane >> 4;
  int row  = m0 + (lane & 15);
  const bf16_t* Ar = A + (size_t)row * lda + k0;
  v16bf a;
#pragma unroll
  for (int r=0;r<8;++r){
    int kb = 2*r + half*16;
    a[2*r]   = Ar[kb];
    a[2*r+1] = Ar[kb+1];
  }
  return a;
}

// B fragment: 32x16 bf16 (K x N), source W[n][k] row-major (B^T).
__device__ __forceinline__ v16bf load_b_frag(const bf16_t* W, int ldw, int n0, int k0){
  int lane = threadIdx.x & 31;
  int half = lane >> 4;
  int n = n0 + (lane & 15);
  const bf16_t* Wr = W + (size_t)n * ldw + k0 + half*16;
  v16bf b;
#pragma unroll
  for (int r=0;r<8;++r){
    b[2*r]   = Wr[2*r];
    b[2*r+1] = Wr[2*r+1];
  }
  return b;
}

__device__ __forceinline__ v8f wmma_bf16(v16bf a, v16bf b, v8f c){
  return __builtin_amdgcn_wmma_f32_16x16x32_bf16(false, a, false, b, (short)0, c, false, false);
}

// ---------------- weight fp32 -> bf16 conversion (8 jobs, one launch) --------
struct ConvJobs {
  const float* src[8];
  bf16_t*      dst[8];
  int          n[8];
};
__global__ void conv8_kernel(ConvJobs jobs){
  int jid = blockIdx.y;
  int i = blockIdx.x * blockDim.x + threadIdx.x;
  if (i < jobs.n[jid]) jobs.dst[jid][i] = (bf16_t)jobs.src[jid][i];
}

// ---------------- embedding gather (+mask) -> bf16 ----------------
// sShift: log2(S) so b/s come from shifts, not integer division
__global__ void embed_kernel(const int* __restrict__ words, const int* __restrict__ idx,
                             const int* __restrict__ imask, const float* __restrict__ table,
                             bf16_t* __restrict__ out, int sShift, int T){
  int b = blockIdx.x >> sShift;
  int s = blockIdx.x & ((1 << sShift) - 1);
  int S = 1 << sShift;
  int tok = idx[b*S + s];
  int w   = words[b*T + tok];
  float mk = (float)imask[b*T + tok];
  const float* src = table + (size_t)w * DD;
  bf16_t* dst = out + ((size_t)b*S + s) * DD;
  for (int d = threadIdx.x; d < DD; d += blockDim.x)
    dst[d] = (bf16_t)(src[d] * mk);
}

// ---------------- input projection GEMM: pre = x @ Wih^T + b ----------------
// Output layout is SCAN-OPTIMIZED: pre[dir][g][s][b]  (batch contiguous).
// grid ((B*S)/16, GG/128, 2), block 256 (8 waves). sShift = log2(S).
__global__ void __launch_bounds__(256) pre_gemm_kernel(
    const bf16_t* __restrict__ X,
    const bf16_t* __restrict__ WihF, const bf16_t* __restrict__ WihB,
    const float* __restrict__ biasF, const float* __restrict__ biasB,
    float* __restrict__ pre, int sShift){
  int dir  = blockIdx.z;
  const bf16_t* W   = dir ? WihB : WihF;
  const float*  bia = dir ? biasB : biasF;
  int wave = threadIdx.x >> 5;
  int lane = threadIdx.x & 31;
  int m0 = blockIdx.x * 16;              // row = b*S + s; tile stays within one b
  int n0 = blockIdx.y * 128 + wave * 16;
  int S  = 1 << sShift;

  v8f acc = zero8();
#pragma unroll 4
  for (int kc = 0; kc < DD/32; ++kc){
    __builtin_prefetch(X + (size_t)(m0 + (lane & 15)) * DD + kc*32 + 64, 0, 1);
    v16bf a = load_a_frag(X, DD, m0, kc*32);
    v16bf b = load_b_frag(W, DD, n0, kc*32);
    acc = wmma_bf16(a, b, acc);
  }
  int half = lane >> 4;
  int col  = n0 + (lane & 15);           // gate-channel g
  float bv = bia[col];
  int bidx  = m0 >> sShift;
  int sbase = (m0 & (S - 1)) + half*8;
  float* preD = pre + (size_t)dir * GG * S * BB;
#pragma unroll
  for (int r=0;r<8;++r){
    // [g][s][b] layout
    preD[((size_t)col * S + (sbase + r)) * BB + bidx] = acc[r] + bv;
  }
}

// ---------------- BiLSTM recurrent scans (word+attr, both dirs, one launch) --
// grid 4: 0=word fwd, 1=word bwd, 2=attr fwd, 3=attr bwd. block 768 (24 waves).
// Wave w owns hidden cols [w*16, w*16+16) for all 4 gates and both 16-row batch
// tiles; cell state c lives in accumulator registers across all timesteps.
__global__ void __launch_bounds__(768, 1) lstm_scan_kernel(
    const float* __restrict__ pre_w, const float* __restrict__ pre_a, // [2][GG][S][B]
    const bf16_t* __restrict__ WhhFw, const bf16_t* __restrict__ WhhBw,
    const bf16_t* __restrict__ WhhFa, const bf16_t* __restrict__ WhhBa,
    const int* __restrict__ wlens, const int* __restrict__ alens,
    float* __restrict__ feats,              // [B*SS][768]   (word only)
    float* __restrict__ afeat){             // [B][768]      (attr only)
  __shared__ bf16_t hsh[BB * HPAD];

  int gb   = blockIdx.x;
  bool isAttr = gb >= 2;
  int dir  = gb & 1;
  int S    = isAttr ? SA : SS;
  const float*  preBase = isAttr ? pre_a : pre_w;
  const float*  preD = preBase + (size_t)dir * GG * S * BB;
  const bf16_t* W = isAttr ? (dir ? WhhBa : WhhFa) : (dir ? WhhBw : WhhFw);
  const int*    lens = isAttr ? alens : wlens;
  float* featOut = isAttr ? nullptr : feats;
  float* hTout   = isAttr ? afeat   : nullptr;

  int tid  = threadIdx.x;
  int wave = tid >> 5;            // 0..23
  int lane = tid & 31;
  int half = lane >> 4;
  int ln   = lane & 15;
  int j0   = wave * 16;

  for (int i = tid; i < BB*HPAD; i += blockDim.x) hsh[i] = (bf16_t)0.0f;

  float ccar[2][8];
  int   lenv[2][8];
#pragma unroll
  for (int m=0;m<2;++m)
#pragma unroll
    for (int r=0;r<8;++r){
      ccar[m][r] = 0.0f;
      lenv[m][r] = lens[m*16 + r + half*8];
    }
  __syncthreads();

  for (int t = 0; t < S; ++t){
    int sp = dir ? (S-1-t) : t;

    // init accumulators from pre: z = pre + h @ Whh^T
    // [g][s][b] layout -> 8 contiguous floats per (m, gate) tile
    v8f acc[2][4];
#pragma unroll
    for (int m=0;m<2;++m)
#pragma unroll
      for (int q=0;q<4;++q){
        const float* pp = preD + ((size_t)(q*HDIM + j0 + ln) * S + sp) * BB + m*16 + half*8;
        v8f c;
#pragma unroll
        for (int r=0;r<8;++r) c[r] = pp[r];
        acc[m][q] = c;
      }

    // prefetch next timestep's C-init rows (one 128B line per gate-channel)
    if (t + 1 < S){
      int spn = dir ? (S-1-(t+1)) : (t+1);
#pragma unroll
      for (int q=0;q<4;++q)
        __builtin_prefetch(preD + ((size_t)(q*HDIM + j0 + ln) * S + spn) * BB, 0, 1);
    }

    // accumulate over K = HDIM: A fragments from LDS h, B from Whh
#pragma unroll 2
    for (int kc = 0; kc < HDIM/32; ++kc){
      v16bf a0 = load_a_frag(hsh, HPAD,  0, kc*32);
      v16bf a1 = load_a_frag(hsh, HPAD, 16, kc*32);
#pragma unroll
      for (int q=0;q<4;++q){
        v16bf b = load_b_frag(W, HDIM, q*HDIM + j0, kc*32);
        acc[0][q] = wmma_bf16(a0, b, acc[0][q]);
        acc[1][q] = wmma_bf16(a1, b, acc[1][q]);
      }
    }

    // gate math + masking (hsh reads happen before the barrier)
    float hkeep[2][8];
#pragma unroll
    for (int m=0;m<2;++m)
#pragma unroll
      for (int r=0;r<8;++r){
        int brow = m*16 + r + half*8;
        float iv = acc[m][0][r];
        float fv = acc[m][1][r];
        float gv = acc[m][2][r];
        float ov = acc[m][3][r];
        float cn = sigmf_(fv)*ccar[m][r] + sigmf_(iv)*tanhf(gv);
        float hn = sigmf_(ov)*tanhf(cn);
        bool msk = sp < lenv[m][r];
        float hp = (float)hsh[brow*HPAD + j0 + ln];
        hkeep[m][r] = msk ? hn : hp;
        ccar[m][r]  = msk ? cn : ccar[m][r];
        if (featOut)
          featOut[((size_t)brow * S + sp) * (2*HDIM) + dir*HDIM + j0 + ln] = msk ? hn : 0.0f;
      }
    __syncthreads();
#pragma unroll
    for (int m=0;m<2;++m)
#pragma unroll
      for (int r=0;r<8;++r){
        int brow = m*16 + r + half*8;
        hsh[brow*HPAD + j0 + ln] = (bf16_t)hkeep[m][r];
      }
    __syncthreads();
  }

  if (hTout){
    for (int i = tid; i < BB*HDIM; i += blockDim.x){
      int b = i / HDIM, j = i % HDIM;
      hTout[(size_t)b * (2*HDIM) + dir*HDIM + j] = (float)hsh[b*HPAD + j];
    }
  }
}

// ---------------- attention + layernorm + tag scores ----------------
__global__ void __launch_bounds__(256) attn_scores_kernel(
    const float* __restrict__ features,   // [B*S][768]
    const float* __restrict__ attr_feat,  // [B][768]
    const int* __restrict__ lens,
    const float* __restrict__ ln_scale, const float* __restrict__ ln_bias, // [1536]
    const float* __restrict__ W_tag, const float* __restrict__ b_tag,      // [30][1536],[30]
    float* __restrict__ all_scores){      // [B*S][32]
  __shared__ float af[2*HDIM];
  __shared__ float ssh[SS];
  __shared__ float ash[SS];
  __shared__ float mx_s, sum_s;

  int b = blockIdx.x;
  int len = lens[b];
  int tid = threadIdx.x;

  for (int i = tid; i < 2*HDIM; i += blockDim.x) af[i] = attr_feat[(size_t)b*(2*HDIM) + i];
  __syncthreads();

  if (tid < SS){
    const float* fr = features + ((size_t)b*SS + tid) * (2*HDIM);
    float d = 0.0f;
    for (int k=0;k<2*HDIM;++k) d += fr[k]*af[k];
    ssh[tid] = (tid < len) ? d : -1e30f;
  }
  __syncthreads();
  if (tid == 0){
    float m = -1e30f;
    for (int s=0;s<SS;++s) m = fmaxf(m, ssh[s]);
    float su = 0.0f;
    for (int s=0;s<SS;++s) su += __expf(ssh[s]-m);
    mx_s = m; sum_s = su;
  }
  __syncthreads();
  if (tid < SS) ash[tid] = __expf(ssh[tid]-mx_s) / sum_s;
  __syncthreads();

  int wv = tid >> 5, lane = tid & 31;
  for (int it = 0; it < SS/8; ++it){
    int sj = it*8 + wv;
    float a = ash[sj];
    const float* fr = features + ((size_t)b*SS + sj) * (2*HDIM);
    float sum = 0.0f, sq = 0.0f;
    for (int k = lane; k < 1536; k += 32){
      float x = (k < 2*HDIM) ? fr[k] : a * af[k - 2*HDIM];
      sum += x; sq += x*x;
    }
#pragma unroll
    for (int off = 16; off > 0; off >>= 1){
      sum += __shfl_xor(sum, off, 32);
      sq  += __shfl_xor(sq,  off, 32);
    }
    float mu  = sum * (1.0f/1536.0f);
    float var = sq * (1.0f/1536.0f) - mu*mu;
    float rs  = rsqrtf(var + 1e-5f);
    if (lane < LLAB){
      const float* wt = W_tag + (size_t)lane * 1536;
      float accL = b_tag[lane];
      for (int k=0;k<1536;++k){
        float x  = (k < 2*HDIM) ? fr[k] : a * af[k - 2*HDIM];
        float xn = (x - mu) * rs * ln_scale[k] + ln_bias[k];
        accL += xn * wt[k];
      }
      all_scores[((size_t)b*SS + sj)*32 + lane] = accL;
    }
  }
}

// ---------------- CRF: logZ, gold, viterbi ----------------
__global__ void __launch_bounds__(32) crf_kernel(
    const float* __restrict__ sc,       // [B*S][32]
    const float* __restrict__ trans,    // [30][30]
    const int* __restrict__ lens,
    const int* __restrict__ labels,     // [B][S]
    float* __restrict__ out){           // [32] nll, [32] best, [B*S] decode
  __shared__ int bps[SS][32];

  int b = blockIdx.x;
  int j = threadIdx.x;
  bool act = j < LLAB;
  int jc = act ? j : 0;
  int len = lens[b];
  const float NEG = -1e30f;
  const float* scb = sc + (size_t)b * SS * 32;

  float alpha = act ? (trans[START_I*LLAB + jc] + scb[jc]) : NEG;
  float v     = alpha;

  for (int t = 1; t < SS; ++t){
    float emit = act ? scb[t*32 + jc] : 0.0f;
    float mxv = NEG;
#pragma unroll 1
    for (int i=0;i<LLAB;++i){
      float ai = __shfl(alpha, i, 32) + trans[i*LLAB + jc];
      mxv = fmaxf(mxv, ai);
    }
    float se = 0.0f;
#pragma unroll 1
    for (int i=0;i<LLAB;++i){
      float ai = __shfl(alpha, i, 32) + trans[i*LLAB + jc];
      se += __expf(ai - mxv);
    }
    float newa = mxv + __logf(se) + emit;
    float best = NEG; int barg = 0;
#pragma unroll 1
    for (int i=0;i<LLAB;++i){
      float cv = __shfl(v, i, 32) + trans[i*LLAB + jc];
      if (cv > best){ best = cv; barg = i; }
    }
    float newv = best + emit;
    bool msk = t < len;
    int bpv = act ? (msk ? barg : j) : j;
    if (act){
      alpha = msk ? newa : alpha;
      v     = msk ? newv : v;
    }
    bps[t][j] = bpv;
  }
  __syncthreads();

  float fz = act ? (alpha + trans[jc*LLAB + END_I]) : NEG;
  float fv = act ? (v     + trans[jc*LLAB + END_I]) : NEG;

  float mz = NEG;
  for (int i=0;i<LLAB;++i) mz = fmaxf(mz, __shfl(fz, i, 32));
  float sz = 0.0f;
  for (int i=0;i<LLAB;++i) sz += __expf(__shfl(fz, i, 32) - mz);
  float logZ = mz + __logf(sz);

  float bbest = NEG; int blast = 0;
  for (int i=0;i<LLAB;++i){
    float bi = __shfl(fv, i, 32);
    if (bi > bbest){ bbest = bi; blast = i; }
  }

  if (j == 0){
    const int* lab = labels + (size_t)b * SS;
    float g = 0.0f;
    for (int t=0;t<SS;++t) if (t < len) g += scb[t*32 + lab[t]];
    for (int t=1;t<SS;++t) if (t < len) g += trans[lab[t-1]*LLAB + lab[t]];
    g += trans[START_I*LLAB + lab[0]];
    g += trans[lab[len-1]*LLAB + END_I];

    out[b]      = logZ - g;
    out[BB + b] = bbest;

    int cur = blast;
    for (int t = SS-1; t >= 0; --t){
      out[2*BB + (size_t)b*SS + t] = (t < len) ? (float)cur : 0.0f;
      if (t > 0) cur = bps[t][cur];
    }
  }
}

// ---------------- host-side launcher ----------------
extern "C" void kernel_launch(void* const* d_in, const int* in_sizes, int n_in,
                              void* d_out, int out_size, void* d_ws, size_t ws_size,
                              hipStream_t stream) {
  const int*   words        = (const int*)  d_in[0];
  const int*   attr_words   = (const int*)  d_in[1];
  const int*   wlens        = (const int*)  d_in[2];
  const int*   alens        = (const int*)  d_in[3];
  const int*   o2t          = (const int*)  d_in[4];
  const int*   ao2t         = (const int*)  d_in[5];
  const int*   imask        = (const int*)  d_in[6];
  const int*   aimask       = (const int*)  d_in[7];
  const int*   labels       = (const int*)  d_in[8];
  const float* table        = (const float*)d_in[9];
  const float* Wih_f_w      = (const float*)d_in[10];
  const float* Whh_f_w      = (const float*)d_in[11];
  const float* b_f_w        = (const float*)d_in[12];
  const float* Wih_b_w      = (const float*)d_in[13];
  const float* Whh_b_w      = (const float*)d_in[14];
  const float* b_b_w        = (const float*)d_in[15];
  const float* Wih_f_a      = (const float*)d_in[16];
  const float* Whh_f_a      = (const float*)d_in[17];
  const float* b_f_a        = (const float*)d_in[18];
  const float* Wih_b_a      = (const float*)d_in[19];
  const float* Whh_b_a      = (const float*)d_in[20];
  const float* b_b_a        = (const float*)d_in[21];
  const float* ln_scale     = (const float*)d_in[22];
  const float* ln_bias      = (const float*)d_in[23];
  const float* W_tag        = (const float*)d_in[24];
  const float* b_tag        = (const float*)d_in[25];
  const float* trans        = (const float*)d_in[26];

  char* w = (char*)d_ws;
  size_t off = 0;
  auto alloc = [&](size_t bytes) -> void* {
    void* p = w + off;
    off = (off + bytes + 255) & ~(size_t)255;
    return p;
  };
  bf16_t* xw_bf  = (bf16_t*)alloc((size_t)BB*SS*DD*2);
  bf16_t* xa_bf  = (bf16_t*)alloc((size_t)BB*SA*DD*2);
  bf16_t* WihFw  = (bf16_t*)alloc((size_t)GG*DD*2);
  bf16_t* WihBw  = (bf16_t*)alloc((size_t)GG*DD*2);
  bf16_t* WhhFw  = (bf16_t*)alloc((size_t)GG*HDIM*2);
  bf16_t* WhhBw  = (bf16_t*)alloc((size_t)GG*HDIM*2);
  bf16_t* WihFa  = (bf16_t*)alloc((size_t)GG*DD*2);
  bf16_t* WihBa  = (bf16_t*)alloc((size_t)GG*DD*2);
  bf16_t* WhhFa  = (bf16_t*)alloc((size_t)GG*HDIM*2);
  bf16_t* WhhBa  = (bf16_t*)alloc((size_t)GG*HDIM*2);
  float*  pre_w  = (float*) alloc((size_t)2*GG*SS*BB*4);   // [dir][g][s][b]
  float*  pre_a  = (float*) alloc((size_t)2*GG*SA*BB*4);   // [dir][g][s][b]
  float*  feats  = (float*) alloc((size_t)BB*SS*(2*HDIM)*4);
  float*  afeat  = (float*) alloc((size_t)BB*(2*HDIM)*4);
  float*  ascore = (float*) alloc((size_t)BB*SS*32*4);
  (void)ws_size; (void)n_in; (void)in_sizes; (void)out_size;

  // 1) weight conversions fp32 -> bf16 (single launch, 8 jobs)
  {
    ConvJobs jobs;
    const float* srcs[8] = {Wih_f_w, Wih_b_w, Whh_f_w, Whh_b_w,
                            Wih_f_a, Wih_b_a, Whh_f_a, Whh_b_a};
    bf16_t*      dsts[8] = {WihFw, WihBw, WhhFw, WhhBw,
                            WihFa, WihBa, WhhFa, WhhBa};
    int          ns[8]   = {GG*DD, GG*DD, GG*HDIM, GG*HDIM,
                            GG*DD, GG*DD, GG*HDIM, GG*HDIM};
    for (int i=0;i<8;++i){ jobs.src[i]=srcs[i]; jobs.dst[i]=dsts[i]; jobs.n[i]=ns[i]; }
    dim3 g((GG*DD + 255)/256, 8);
    conv8_kernel<<<g, 256, 0, stream>>>(jobs);
  }

  // 2) embedding gathers (sShift = log2(S))
  embed_kernel<<<BB*SS, 256, 0, stream>>>(words,      o2t,  imask,  table, xw_bf, 7, TT);
  embed_kernel<<<BB*SA, 256, 0, stream>>>(attr_words, ao2t, aimask, table, xa_bf, 5, TA);

  // 3) input projection GEMMs (WMMA bf16), scan-optimized output layout
  {
    dim3 gw(BB*SS/16, GG/128, 2);
    pre_gemm_kernel<<<gw, 256, 0, stream>>>(xw_bf, WihFw, WihBw, b_f_w, b_b_w, pre_w, 7);
    dim3 ga(BB*SA/16, GG/128, 2);
    pre_gemm_kernel<<<ga, 256, 0, stream>>>(xa_bf, WihFa, WihBa, b_f_a, b_b_a, pre_a, 5);
  }

  // 4) all four recurrent scans concurrently (word f/b + attr f/b)
  lstm_scan_kernel<<<4, 768, 0, stream>>>(pre_w, pre_a, WhhFw, WhhBw, WhhFa, WhhBa,
                                          wlens, alens, feats, afeat);

  // 5) attention + layernorm + tag scores
  attn_scores_kernel<<<BB, 256, 0, stream>>>(feats, afeat, wlens, ln_scale, ln_bias,
                                             W_tag, b_tag, ascore);

  // 6) CRF logZ / gold / viterbi
  crf_kernel<<<BB, 32, 0, stream>>>(ascore, trans, wlens, labels, (float*)d_out);
}
